// majoritySNN_21492016349611
// MI455X (gfx1250) — matrix operand order repaired and checked by reading, not compiled, not run on hardware
//
#include <hip/hip_runtime.h>
#include <hip/hip_bf16.h>
#include <math.h>

typedef __attribute__((ext_vector_type(16))) _Float16 v16h;
typedef __attribute__((ext_vector_type(8)))  _Float16 v8h;
typedef __attribute__((ext_vector_type(8)))  float    v8f;

// ---------------- problem constants ----------------
#define T_STEPS 15
#define C_IN    6
#define H_IN    256
#define W_IN    256
#define C1_OUT  30          // conv1 out channels
#define K1      5           // conv1 kernel
#define C1_K    150         // 6*5*5
#define C1_KP   160         // padded to 5 x 32
#define H_P     128         // pooled H
#define W_P     128
#define H_PP    130         // pooled+padded
#define W_PP    130
#define C2_OUT  250
#define K2K     3
#define C2_K    270         // 30*3*3
#define C2_KP   288         // padded to 9 x 32
#define THR1    15.0f
#define THR2    10.0f
#define KWTA    8
#define RADIUS  1

#define POT_ELEMS   (T_STEPS * C2_OUT * H_P * W_P)   // 61,440,000
#define CHW_ELEMS   (C2_OUT * H_P * W_P)             // 4,096,000
#define HW_ELEMS    (H_P * W_P)                      // 16,384
#define GW1_BLOCKS  (CHW_ELEMS / 256)                // 16,000
#define AM_BLOCKS   1024

// =====================================================================
// conv1 + fire : x (15,6,256,256) f32  ->  spk1 (15,30,256,256) f16 {0,1}
// Per block: 64 spatial (along W) x 32 out-channels, K = 160 (5 x 32)
// Phase 1: im2col into LDS sA[m][k], weights into LDS sW[n][k]
// Phase 2: pure ds_load_b128 + v_wmma_f32_16x16x32_f16
// =====================================================================
__global__ __launch_bounds__(256)
void conv1_fire_kernel(const float* __restrict__ x,
                       const float* __restrict__ w1,
                       _Float16* __restrict__ spk1) {
    const int w0   = blockIdx.x * 64;   // 0,64,128,192
    const int h    = blockIdx.y;        // 0..255
    const int t    = blockIdx.z;        // 0..14
    const int tid  = threadIdx.x;
    const int lane = tid & 31;
    const int wave = tid >> 5;
    const int mWave = (wave & 3) * 16;  // spatial sub-tile
    const int nWave = (wave >> 2) * 16; // channel sub-tile

    __shared__ __align__(32) _Float16 sA[64 * C1_KP]; // [m][k] 20 KB
    __shared__ __align__(32) _Float16 sW[32 * C1_KP]; // [n][k] 10 KB

    // ---- stage weights (f32 -> f16), zero-padded ----
    for (int i = tid; i < 32 * C1_KP; i += 256) {
        int n = i / C1_KP, k = i - n * C1_KP;
        float v = 0.0f;
        if (k < C1_K && n < C1_OUT) {
            int c = k / 25, rem = k % 25, dy = rem / 5, dx = rem % 5;
            v = w1[((n * C_IN + c) * K1 + dy) * K1 + dx];
        }
        sW[i] = (_Float16)v;
    }
    // ---- stage im2col A tile (coalesced over m) ----
    for (int i = tid; i < C1_KP * 64; i += 256) {
        int k = i >> 6, m = i & 63;
        float v = 0.0f;
        if (k < C1_K) {
            int c = k / 25, rem = k % 25, dy = rem / 5, dx = rem % 5;
            int hh = h + dy - 2;
            int ww = w0 + m + dx - 2;
            if (hh >= 0 && hh < H_IN && ww >= 0 && ww < W_IN)
                v = x[((t * C_IN + c) * H_IN + hh) * W_IN + ww];
        }
        sA[m * C1_KP + k] = (_Float16)v;
    }
    __syncthreads();

    // ---- WMMA phase ----
    const int mlane = lane & 15;
    const _Float16* arow = &sA[(mWave + mlane) * C1_KP];
    const _Float16* brow = &sW[(nWave + mlane) * C1_KP];
    const int kHiA = (lane >> 4) * 16;   // A: lane-lo K 0..15, lane-hi 16..31
    const int kHiB = (lane >> 4) * 16;   // B: lane-lo K 0..15, lane-hi 16..31

    v8f acc = {};
#pragma unroll
    for (int kk = 0; kk < C1_KP / 32; ++kk) {
        v8h a0 = *(const v8h*)(arow + kk * 32 + kHiA);
        v8h a1 = *(const v8h*)(arow + kk * 32 + 8 + kHiA);
        v16h a = __builtin_shufflevector(a0, a1, 0, 1, 2, 3, 4, 5, 6, 7,
                                         8, 9, 10, 11, 12, 13, 14, 15);
        v16h b = *(const v16h*)(brow + kk * 32 + kHiB);
        acc = __builtin_amdgcn_wmma_f32_16x16x32_f16(false, a, false, b,
                                                     (short)0, acc, false, false);
    }

    // ---- fire + store spikes (binary, f16) ----
#pragma unroll
    for (int r = 0; r < 8; ++r) {
        int m = mWave + r + ((lane >> 4) << 3);   // spatial within 64-tile
        int n = nWave + (lane & 15);              // out channel within 32
        if (n < C1_OUT) {
            float p = acc[r];
            spk1[((t * C1_OUT + n) * H_IN + h) * W_IN + (w0 + m)] =
                (_Float16)((p > THR1) ? 1.0f : 0.0f);
        }
    }
}

// =====================================================================
// maxpool 2x2 stride 2 + pad 1 : spk1 -> spk_in (15,30,130,130) f16
// =====================================================================
__global__ void maxpool_pad_kernel(const _Float16* __restrict__ spk1,
                                   _Float16* __restrict__ spk_in) {
    long long idx = (long long)blockIdx.x * blockDim.x + threadIdx.x;
    const long long N = (long long)T_STEPS * C1_OUT * H_PP * W_PP;
    if (idx >= N) return;
    int wp = (int)(idx % W_PP);
    int hp = (int)((idx / W_PP) % H_PP);
    int c  = (int)((idx / (W_PP * H_PP)) % C1_OUT);
    int t  = (int)(idx / ((long long)W_PP * H_PP * C1_OUT));
    float v = 0.0f;
    if (hp >= 1 && hp <= H_P && wp >= 1 && wp <= W_P) {
        int h0 = (hp - 1) * 2, w0c = (wp - 1) * 2;
        long long base = ((long long)(t * C1_OUT + c) * H_IN + h0) * W_IN + w0c;
        float a = (float)spk1[base];
        float b = (float)spk1[base + 1];
        float d = (float)spk1[base + W_IN];
        float e = (float)spk1[base + W_IN + 1];
        v = fmaxf(fmaxf(a, b), fmaxf(d, e));
    }
    spk_in[idx] = (_Float16)v;
}

// =====================================================================
// conv2 + fire : spk_in (15,30,130,130) f16 -> pot (15,250,128,128) f32
// Per block: 64 spatial x 32 out-channels, K = 288 (9 x 32)
// =====================================================================
__global__ __launch_bounds__(256)
void conv2_fire_kernel(const _Float16* __restrict__ spk_in,
                       const float* __restrict__ w2,
                       float* __restrict__ pot) {
    const int w0    = blockIdx.x * 64;  // 0,64
    const int h     = blockIdx.y;       // 0..127
    const int z     = blockIdx.z;       // t*8 + channel-group
    const int t     = z >> 3;
    const int nBase = (z & 7) * 32;     // 0..224
    const int tid   = threadIdx.x;
    const int lane  = tid & 31;
    const int wave  = tid >> 5;
    const int mWave = (wave & 3) * 16;
    const int nWave = (wave >> 2) * 16;

    __shared__ __align__(32) _Float16 sA[64 * C2_KP]; // [m][k] 36 KB
    __shared__ __align__(32) _Float16 sW[32 * C2_KP]; // [n][k] 18 KB

    // ---- stage weights ----
    for (int i = tid; i < 32 * C2_KP; i += 256) {
        int n = i / C2_KP, k = i - n * C2_KP;
        int gn = nBase + n;
        float v = 0.0f;
        if (k < C2_K && gn < C2_OUT) {
            int c = k / 9, rem = k % 9, dy = rem / 3, dx = rem % 3;
            v = w2[((gn * C1_OUT + c) * K2K + dy) * K2K + dx];
        }
        sW[i] = (_Float16)v;
    }
    // ---- stage im2col A tile (spk_in is pre-padded: always in bounds) ----
    for (int i = tid; i < C2_KP * 64; i += 256) {
        int k = i >> 6, m = i & 63;
        _Float16 v = (_Float16)0.0f;
        if (k < C2_K) {
            int c = k / 9, rem = k % 9, dy = rem / 3, dx = rem % 3;
            v = spk_in[((t * C1_OUT + c) * H_PP + (h + dy)) * W_PP + (w0 + m + dx)];
        }
        sA[m * C2_KP + k] = v;
    }
    __syncthreads();

    // ---- WMMA phase ----
    const int mlane = lane & 15;
    const _Float16* arow = &sA[(mWave + mlane) * C2_KP];
    const _Float16* brow = &sW[(nWave + mlane) * C2_KP];
    const int kHiA = (lane >> 4) * 16;
    const int kHiB = (lane >> 4) * 16;

    v8f acc = {};
#pragma unroll
    for (int kk = 0; kk < C2_KP / 32; ++kk) {
        v8h a0 = *(const v8h*)(arow + kk * 32 + kHiA);
        v8h a1 = *(const v8h*)(arow + kk * 32 + 8 + kHiA);
        v16h a = __builtin_shufflevector(a0, a1, 0, 1, 2, 3, 4, 5, 6, 7,
                                         8, 9, 10, 11, 12, 13, 14, 15);
        v16h b = *(const v16h*)(brow + kk * 32 + kHiB);
        acc = __builtin_amdgcn_wmma_f32_16x16x32_f16(false, a, false, b,
                                                     (short)0, acc, false, false);
    }

    // ---- fire + store pot (thresholded) ----
#pragma unroll
    for (int r = 0; r < 8; ++r) {
        int m = mWave + r + ((lane >> 4) << 3);
        int n = nBase + nWave + (lane & 15);
        if (n < C2_OUT) {
            float p = acc[r];
            pot[((long long)(t * C2_OUT + n) * H_P + h) * W_P + (w0 + m)] =
                (p > THR2) ? p : 0.0f;
        }
    }
}

// =====================================================================
// pointwise inhibition, pass 1: per-pixel winner channel + spiked flag
// =====================================================================
__global__ void inhibit_winners_kernel(const float* __restrict__ pot,
                                       int* __restrict__ winner,
                                       float* __restrict__ spiked) {
    int hw = blockIdx.x * blockDim.x + threadIdx.x;
    if (hw >= HW_ELEMS) return;
    int args[T_STEPS];
    int nspike = 0;
    int lastClamp = 0;
    for (int t = 0; t < T_STEPS; ++t) {
        const float* p = pot + (long long)t * CHW_ELEMS + hw;
        float mv = p[0];
        int   arg = 0;
        for (int c = 1; c < C2_OUT; ++c) {
            float v = p[(long long)c * HW_ELEMS];
            if (v > mv) { mv = v; arg = c; }
        }
        args[t] = arg;
        int clamp = (mv > 0.0f) ? 1 : 0;
        nspike += clamp;
        lastClamp = clamp;
    }
    int first_t = T_STEPS - nspike;
    if (first_t < 0) first_t = 0;
    if (first_t > T_STEPS - 1) first_t = T_STEPS - 1;
    winner[hw] = args[first_t];
    spiked[hw] = (float)lastClamp;
}

// pass 2: apply coef, write inhibited pot and spk = sign(pot)
__global__ void inhibit_apply_kernel(float* __restrict__ pot,
                                     float* __restrict__ spk,
                                     const int* __restrict__ winner,
                                     const float* __restrict__ spiked) {
    long long idx = (long long)blockIdx.x * blockDim.x + threadIdx.x;
    if (idx >= (long long)POT_ELEMS) return;
    int hw = (int)(idx % HW_ELEMS);
    int c  = (int)((idx / HW_ELEMS) % C2_OUT);
    float coef = (c == winner[hw]) ? spiked[hw] : 0.0f;
    float p = pot[idx] * coef;
    pot[idx] = p;
    spk[idx] = (p > 0.0f) ? 1.0f : 0.0f;
}

// =====================================================================
// k-winners: stage 1 — per (c,h,w): n spikes, value at first_t,
// block-max of candidate value (for the global "v" offset)
// =====================================================================
__global__ __launch_bounds__(256)
void gw_prep_kernel(const float* __restrict__ pot,
                    const float* __restrict__ spk,
                    float* __restrict__ nArr,
                    float* __restrict__ vArr,
                    float* __restrict__ partialMax) {
    __shared__ float red[256];
    int idx = blockIdx.x * 256 + threadIdx.x;   // exactly CHW_ELEMS threads
    float n = 0.0f;
    for (int t = 0; t < T_STEPS; ++t)
        n += spk[(long long)t * CHW_ELEMS + idx];
    int first_t = T_STEPS - (int)n;
    if (first_t < 0) first_t = 0;
    if (first_t > T_STEPS - 1) first_t = T_STEPS - 1;
    float val = pot[(long long)first_t * CHW_ELEMS + idx];
    nArr[idx] = n;
    vArr[idx] = val;
    float cand = (n > 0.0f) ? val : 0.0f;

    red[threadIdx.x] = cand;
    __syncthreads();
    for (int s = 128; s > 0; s >>= 1) {
        if (threadIdx.x < s)
            red[threadIdx.x] = fmaxf(red[threadIdx.x], red[threadIdx.x + s]);
        __syncthreads();
    }
    if (threadIdx.x == 0) partialMax[blockIdx.x] = red[0];
}

// stage 2 — reduce partial maxima, v = max * T
__global__ void gw_vmax_kernel(const float* __restrict__ partialMax,
                               float* __restrict__ vScalar) {
    __shared__ float red[256];
    float m = 0.0f;
    for (int i = threadIdx.x; i < GW1_BLOCKS; i += 256)
        m = fmaxf(m, partialMax[i]);
    red[threadIdx.x] = m;
    __syncthreads();
    for (int s = 128; s > 0; s >>= 1) {
        if (threadIdx.x < s)
            red[threadIdx.x] = fmaxf(red[threadIdx.x], red[threadIdx.x + s]);
        __syncthreads();
    }
    if (threadIdx.x == 0) vScalar[0] = red[0] * (float)T_STEPS;
}

// stage 3 — total = n * (val + v)
__global__ void gw_total_kernel(const float* __restrict__ nArr,
                                const float* __restrict__ vArr,
                                const float* __restrict__ vScalar,
                                float* __restrict__ total) {
    int idx = blockIdx.x * 256 + threadIdx.x;
    total[idx] = nArr[idx] * (vArr[idx] + vScalar[0]);
}

// ------------- argmax over total (two-level, first-max tie rule) -------
__global__ __launch_bounds__(256)
void argmax_part_kernel(const float* __restrict__ total,
                        float* __restrict__ pVal, int* __restrict__ pIdx) {
    __shared__ float sv[256];
    __shared__ int   si[256];
    float best = -1.0f; int bidx = 0x7fffffff;
    for (int i = blockIdx.x * 256 + threadIdx.x; i < CHW_ELEMS; i += AM_BLOCKS * 256) {
        float v = total[i];
        if (v > best || (v == best && i < bidx)) { best = v; bidx = i; }
    }
    sv[threadIdx.x] = best; si[threadIdx.x] = bidx;
    __syncthreads();
    for (int s = 128; s > 0; s >>= 1) {
        if (threadIdx.x < s) {
            float v = sv[threadIdx.x + s]; int j = si[threadIdx.x + s];
            if (v > sv[threadIdx.x] || (v == sv[threadIdx.x] && j < si[threadIdx.x])) {
                sv[threadIdx.x] = v; si[threadIdx.x] = j;
            }
        }
        __syncthreads();
    }
    if (threadIdx.x == 0) { pVal[blockIdx.x] = sv[0]; pIdx[blockIdx.x] = si[0]; }
}

__global__ void argmax_final_kernel(const float* __restrict__ pVal,
                                    const int* __restrict__ pIdx,
                                    int* __restrict__ winners_out, int k,
                                    int* __restrict__ wslot) {
    __shared__ float sv[256];
    __shared__ int   si[256];
    float best = -1.0f; int bidx = 0x7fffffff;
    for (int i = threadIdx.x; i < AM_BLOCKS; i += 256) {
        float v = pVal[i]; int j = pIdx[i];
        if (v > best || (v == best && j < bidx)) { best = v; bidx = j; }
    }
    sv[threadIdx.x] = best; si[threadIdx.x] = bidx;
    __syncthreads();
    for (int s = 128; s > 0; s >>= 1) {
        if (threadIdx.x < s) {
            float v = sv[threadIdx.x + s]; int j = si[threadIdx.x + s];
            if (v > sv[threadIdx.x] || (v == sv[threadIdx.x] && j < si[threadIdx.x])) {
                sv[threadIdx.x] = v; si[threadIdx.x] = j;
            }
        }
        __syncthreads();
    }
    if (threadIdx.x == 0) {
        int idx = si[0];
        int valid = (sv[0] != 0.0f) ? 1 : 0;
        int c = idx / HW_ELEMS;
        int h = (idx % HW_ELEMS) / W_P;
        int w = idx % W_P;
        winners_out[k * 3 + 0] = valid ? c : -1;
        winners_out[k * 3 + 1] = valid ? h : -1;
        winners_out[k * 3 + 2] = valid ? w : -1;
        wslot[0] = c; wslot[1] = h; wslot[2] = w; wslot[3] = valid;
    }
}

__global__ void suppress_kernel(float* __restrict__ total,
                                const int* __restrict__ wslot) {
    if (!wslot[3]) return;
    int idx = blockIdx.x * 256 + threadIdx.x;
    int wc = wslot[0], wh = wslot[1], ww = wslot[2];
    int c = idx / HW_ELEMS;
    int h = (idx % HW_ELEMS) / W_P;
    int w = idx % W_P;
    bool mask = (c == wc) ||
                ((abs(h - wh) <= RADIUS) && (abs(w - ww) <= RADIUS));
    if (mask) total[idx] = 0.0f;
}

// =====================================================================
extern "C" void kernel_launch(void* const* d_in, const int* in_sizes, int n_in,
                              void* d_out, int out_size, void* d_ws, size_t ws_size,
                              hipStream_t stream) {
    const float* x  = (const float*)d_in[0];   // (15,6,256,256)
    const float* w1 = (const float*)d_in[1];   // (30,6,5,5)
    const float* w2 = (const float*)d_in[2];   // (250,30,3,3)

    float* spk = (float*)d_out;                    // (15,250,128,128)
    float* pot = spk + (long long)POT_ELEMS;       // (15,250,128,128)
    int*   win = (int*)(pot + (long long)POT_ELEMS); // (8,3) int32 bits

    // ---- workspace layout ----
    char* p = (char*)d_ws;
    auto take = [&](size_t bytes) {
        char* r = p;
        p += (bytes + 255) & ~(size_t)255;
        return (void*)r;
    };
    _Float16* spk1    = (_Float16*)take((size_t)T_STEPS * C1_OUT * H_IN * W_IN * 2);
    _Float16* spk_in  = (_Float16*)take((size_t)T_STEPS * C1_OUT * H_PP * W_PP * 2);
    int*      winner  = (int*)     take((size_t)HW_ELEMS * 4);
    float*    spiked  = (float*)   take((size_t)HW_ELEMS * 4);
    float*    nArr    = (float*)   take((size_t)CHW_ELEMS * 4);
    float*    vArr    = (float*)   take((size_t)CHW_ELEMS * 4);
    float*    total   = (float*)   take((size_t)CHW_ELEMS * 4);
    float*    partMax = (float*)   take((size_t)GW1_BLOCKS * 4);
    float*    vScalar = (float*)   take(256);
    float*    pVal    = (float*)   take((size_t)AM_BLOCKS * 4);
    int*      pIdx    = (int*)     take((size_t)AM_BLOCKS * 4);
    int*      wslot   = (int*)     take(256);

    // layer 1: conv + fire (WMMA f16 -> f32)
    conv1_fire_kernel<<<dim3(W_IN / 64, H_IN, T_STEPS), 256, 0, stream>>>(x, w1, spk1);

    // maxpool + pad
    {
        long long N = (long long)T_STEPS * C1_OUT * H_PP * W_PP;
        maxpool_pad_kernel<<<(unsigned)((N + 255) / 256), 256, 0, stream>>>(spk1, spk_in);
    }

    // layer 2: conv + fire (WMMA f16 -> f32), writes pot into d_out
    conv2_fire_kernel<<<dim3(W_P / 64, H_P, T_STEPS * 8), 256, 0, stream>>>(spk_in, w2, pot);

    // pointwise inhibition
    inhibit_winners_kernel<<<(HW_ELEMS + 255) / 256, 256, 0, stream>>>(pot, winner, spiked);
    inhibit_apply_kernel<<<(unsigned)(((long long)POT_ELEMS + 255) / 256), 256, 0, stream>>>(
        pot, spk, winner, spiked);

    // winner totals
    gw_prep_kernel<<<GW1_BLOCKS, 256, 0, stream>>>(pot, spk, nArr, vArr, partMax);
    gw_vmax_kernel<<<1, 256, 0, stream>>>(partMax, vScalar);
    gw_total_kernel<<<GW1_BLOCKS, 256, 0, stream>>>(nArr, vArr, vScalar, total);

    // k-WTA with radius-1 lateral suppression
    for (int k = 0; k < KWTA; ++k) {
        argmax_part_kernel<<<AM_BLOCKS, 256, 0, stream>>>(total, pVal, pIdx);
        argmax_final_kernel<<<1, 256, 0, stream>>>(pVal, pIdx, win, k, wslot);
        suppress_kernel<<<GW1_BLOCKS, 256, 0, stream>>>(total, wslot);
    }
}
